// TimeSeriesEncoder_17016660427569
// MI455X (gfx1250) — matrix operand order repaired and checked by reading, not compiled, not run on hardware
//
#include <hip/hip_runtime.h>
#include <hip/hip_bf16.h>

// ---------------------------------------------------------------------------
// TimeSeriesEncoder for MI455X (gfx1250, wave32, WMMA).
// All GEMMs (QKV / O / FF1 / FF2) and both attention GEMMs run on
// v_wmma_f32_16x16x32_bf16 with fp32 accumulation.
// ---------------------------------------------------------------------------

typedef __attribute__((ext_vector_type(8)))  float  v8f;
typedef __attribute__((ext_vector_type(8)))  __bf16 v8bf;
typedef __attribute__((ext_vector_type(16))) __bf16 v16bf;

#define NBATCH 8
#define SEQ    1024
#define EMB    512
#define FFD    2048
#define NLAY   6
#define NHEAD  8
#define HDIM   64
#define MTOK   (NBATCH * SEQ)   // 8192 tokens

__device__ __forceinline__ __bf16 to_bf16(float f) {
  unsigned u = __builtin_bit_cast(unsigned, f);
  u += 0x7FFFu + ((u >> 16) & 1u);          // round-to-nearest-even
  unsigned short hs = (unsigned short)(u >> 16);
  return __builtin_bit_cast(__bf16, hs);
}

// ---------------------------------------------------------------------------
// h = x_f32 @ W_enc + b_enc + emb_hour[x3] + emb_weekday[x2] + emb_day[x1]
//     + emb_month[x0]  -> fp32 + bf16 copies
// ---------------------------------------------------------------------------
__global__ void embed_kernel(const int* __restrict__ x, const float* __restrict__ Wenc,
                             const float* __restrict__ benc, const float* __restrict__ em,
                             const float* __restrict__ ed, const float* __restrict__ ew,
                             const float* __restrict__ eh, float* __restrict__ h32,
                             __bf16* __restrict__ hbf) {
  int t = blockIdx.x;                      // token index [0, 8192)
  int xd[8];
#pragma unroll
  for (int d = 0; d < 8; ++d) xd[d] = x[t * 8 + d];
  for (int e = threadIdx.x; e < EMB; e += 256) {
    float v = benc[e];
#pragma unroll
    for (int d = 0; d < 8; ++d) v += (float)xd[d] * Wenc[d * EMB + e];
    v += eh[xd[3] * EMB + e] + ew[xd[2] * EMB + e] +
         ed[xd[1] * EMB + e] + em[xd[0] * EMB + e];
    size_t off = (size_t)t * EMB + e;
    h32[off] = v;
    hbf[off] = to_bf16(v);
  }
}

// ---------------------------------------------------------------------------
// Convert fp32 weight W[K, Nw] (row-major) into bf16 B^T [Nw, K] so GEMM
// B-fragments become contiguous 32B loads.
// ---------------------------------------------------------------------------
__global__ void transpose_bf16(const float* __restrict__ W, __bf16* __restrict__ BT,
                               int K, int Nw) {
  int idx = blockIdx.x * 256 + threadIdx.x;
  if (idx >= K * Nw) return;
  int kk = idx / Nw;
  int n  = idx - kk * Nw;
  BT[(size_t)n * K + kk] = to_bf16(W[idx]);
}

// v [N,S,E] bf16  ->  vT [N, H, HD, S] bf16 (per-head transposed for P@V WMMA)
__global__ void head_transpose(const __bf16* __restrict__ v, __bf16* __restrict__ vT) {
  int idx = blockIdx.x * 256 + threadIdx.x;     // < N*S*E = 4M
  int e = idx & (EMB - 1);
  int s = (idx >> 9) & (SEQ - 1);
  int n = idx >> 19;
  int hh = e >> 6, d = e & 63;
  vT[((((size_t)n * NHEAD + hh) * HDIM + d) << 10) + s] = v[idx];
}

// ---------------------------------------------------------------------------
// C[M,Nw] = A[M,K](bf16, row-major) * BT[Nw,K](bf16) + bias; optional ReLU.
// One wave computes a 16x64 tile; 8 waves/block => 128x64 block tile.
// ---------------------------------------------------------------------------
__global__ void gemm_bf16(const __bf16* __restrict__ A, const __bf16* __restrict__ BT,
                          const float* __restrict__ bias, float* __restrict__ Cf,
                          __bf16* __restrict__ Cb, int M, int K, int Nw, int relu) {
  int lane = threadIdx.x & 31;
  int wv   = threadIdx.x >> 5;
  int m0 = blockIdx.y * 128 + wv * 16;
  int n0 = blockIdx.x * 64;
  int lm = lane & 15;
  bool lh = lane < 16;
  int kbA = lh ? 0 : 16;   // A-fragment K base (contiguous k = 16h + s)
  int ksB = lh ? 0 : 16;   // B-fragment K base (contiguous k = 16h + s, same as A)

  v8f acc[4];
#pragma unroll
  for (int nt = 0; nt < 4; ++nt)
#pragma unroll
    for (int r = 0; r < 8; ++r) acc[nt][r] = 0.f;

  const __bf16* arow = A + (size_t)(m0 + lm) * K;
  for (int kc = 0; kc < K; kc += 32) {
    v8bf lo = *(const v8bf*)(arow + kc + kbA);
    v8bf hi = *(const v8bf*)(arow + kc + kbA + 8);
    v16bf a;
#pragma unroll
    for (int i = 0; i < 8; ++i) { a[i] = lo[i]; a[i + 8] = hi[i]; }
#pragma unroll
    for (int nt = 0; nt < 4; ++nt) {
      const __bf16* bp = BT + (size_t)(n0 + nt * 16 + lm) * K + kc + ksB;
      v16bf b = *(const v16bf*)bp;
      acc[nt] = __builtin_amdgcn_wmma_f32_16x16x32_bf16(
          false, a, false, b, (short)0, acc[nt], false, false);
    }
  }

#pragma unroll
  for (int nt = 0; nt < 4; ++nt) {
    int nc = n0 + nt * 16 + lm;
    float bv = bias ? bias[nc] : 0.f;
#pragma unroll
    for (int r = 0; r < 8; ++r) {
      int m = m0 + r + (lh ? 0 : 8);
      float v = acc[nt][r] + bv;
      if (relu) v = fmaxf(v, 0.f);
      size_t off = (size_t)m * Nw + nc;
      if (Cf) Cf[off] = v;
      if (Cb) Cb[off] = to_bf16(v);
    }
  }
}

// ---------------------------------------------------------------------------
// Flash-style attention, one wave per (batch n, head hh, 16-query tile).
// Scores: Q(16x64) @ K^T via 2 WMMA/16-key tile; online softmax over 32-key
// blocks; P(16x32) transposed through LDS into A-layout; P@V via WMMA with
// pre-transposed V.  Reference semantics: mask==0 -> -1e30, scale 1/sqrt(E).
// ---------------------------------------------------------------------------
__global__ void attn_kernel(const __bf16* __restrict__ q, const __bf16* __restrict__ k,
                            const __bf16* __restrict__ vT, const int* __restrict__ mask,
                            __bf16* __restrict__ o) {
  __shared__ float pbuf[16 * 32];
  int lane = threadIdx.x;
  int q0 = blockIdx.x * 16, hh = blockIdx.y, n = blockIdx.z;
  int lm = lane & 15;
  bool lh = lane < 16;
  int kbA = lh ? 0 : 16;
  int ksB = lh ? 0 : 16;
  const float inv_scale = 0.044194173824159216f;   // 1/sqrt(512)

  // preload Q fragments (two K-chunks covering head dim 64)
  v16bf aq[2];
  {
    const __bf16* qb = q + ((size_t)(n * SEQ + q0 + lm)) * EMB + hh * HDIM;
#pragma unroll
    for (int c = 0; c < 2; ++c) {
      v8bf lo = *(const v8bf*)(qb + c * 32 + kbA);
      v8bf hi = *(const v8bf*)(qb + c * 32 + kbA + 8);
#pragma unroll
      for (int i = 0; i < 8; ++i) { aq[c][i] = lo[i]; aq[c][i + 8] = hi[i]; }
    }
  }

  v8f oacc[4];
#pragma unroll
  for (int nt = 0; nt < 4; ++nt)
#pragma unroll
    for (int r = 0; r < 8; ++r) oacc[nt][r] = 0.f;
  float rmax[8], rsum[8];
#pragma unroll
  for (int r = 0; r < 8; ++r) { rmax[r] = -1e38f; rsum[r] = 0.f; }

  for (int j = 0; j < SEQ / 32; ++j) {
    int kp0 = j * 32;
    v8f e0, e1;
#pragma unroll
    for (int r = 0; r < 8; ++r) { e0[r] = 0.f; e1[r] = 0.f; }
#pragma unroll
    for (int c = 0; c < 2; ++c) {
      const __bf16* kb0 = k + ((size_t)(n * SEQ + kp0 + lm)) * EMB + hh * HDIM + c * 32 + ksB;
      const __bf16* kb1 = k + ((size_t)(n * SEQ + kp0 + 16 + lm)) * EMB + hh * HDIM + c * 32 + ksB;
      v16bf b0 = *(const v16bf*)kb0;
      v16bf b1 = *(const v16bf*)kb1;
      e0 = __builtin_amdgcn_wmma_f32_16x16x32_bf16(false, aq[c], false, b0, (short)0, e0, false, false);
      e1 = __builtin_amdgcn_wmma_f32_16x16x32_bf16(false, aq[c], false, b1, (short)0, e1, false, false);
    }

    int mk0 = mask[n * SEQ + kp0 + lm];
    int mk1 = mask[n * SEQ + kp0 + 16 + lm];
    float p0[8], p1[8], cr[8];
#pragma unroll
    for (int r = 0; r < 8; ++r) {
      float es0 = mk0 ? e0[r] * inv_scale : -1e30f;
      float es1 = mk1 ? e1[r] * inv_scale : -1e30f;
      float mx = fmaxf(es0, es1);
#pragma unroll
      for (int d = 1; d < 16; d <<= 1) mx = fmaxf(mx, __shfl_xor(mx, d, 32));
      float nm = fmaxf(rmax[r], mx);
      cr[r] = __expf(rmax[r] - nm);
      p0[r] = __expf(es0 - nm);
      p1[r] = __expf(es1 - nm);
      float sm = p0[r] + p1[r];
#pragma unroll
      for (int d = 1; d < 16; d <<= 1) sm += __shfl_xor(sm, d, 32);
      rsum[r] = rsum[r] * cr[r] + sm;
      rmax[r] = nm;
    }
#pragma unroll
    for (int nt = 0; nt < 4; ++nt)
#pragma unroll
      for (int r = 0; r < 8; ++r) oacc[nt][r] *= cr[r];

    // transpose P (C-layout -> A-layout) through LDS
    int mrow = lh ? 0 : 8;
#pragma unroll
    for (int r = 0; r < 8; ++r) {
      pbuf[(r + mrow) * 32 + lm]      = p0[r];
      pbuf[(r + mrow) * 32 + 16 + lm] = p1[r];
    }
    __syncthreads();
    v16bf pa;
#pragma unroll
    for (int h2 = 0; h2 < 8; ++h2) {
      pa[h2]     = to_bf16(pbuf[lm * 32 + kbA + h2]);
      pa[h2 + 8] = to_bf16(pbuf[lm * 32 + kbA + 8 + h2]);
    }
    __syncthreads();

#pragma unroll
    for (int nt = 0; nt < 4; ++nt) {
      const __bf16* vp = vT + ((((size_t)n * NHEAD + hh) * HDIM) + nt * 16 + lm) * SEQ + kp0 + ksB;
      v16bf bv = *(const v16bf*)vp;
      oacc[nt] = __builtin_amdgcn_wmma_f32_16x16x32_bf16(
          false, pa, false, bv, (short)0, oacc[nt], false, false);
    }
  }

#pragma unroll
  for (int nt = 0; nt < 4; ++nt)
#pragma unroll
    for (int r = 0; r < 8; ++r) {
      int m = q0 + r + (lh ? 0 : 8);
      float val = oacc[nt][r] / rsum[r];
      o[((size_t)(n * SEQ + m)) * EMB + hh * HDIM + nt * 16 + lm] = to_bf16(val);
    }
}

// ---------------------------------------------------------------------------
// out = LayerNorm(a + res) * g + b   (E = 512 per token, eps = 1e-5)
// ---------------------------------------------------------------------------
__global__ void add_ln_kernel(const float* __restrict__ a, const float* __restrict__ res,
                              const float* __restrict__ g, const float* __restrict__ bta,
                              float* __restrict__ out32, __bf16* __restrict__ outbf) {
  __shared__ float red[256];
  int t = blockIdx.x;
  int e = threadIdx.x;
  size_t base = (size_t)t * EMB;
  float v0 = a[base + e]       + res[base + e];
  float v1 = a[base + e + 256] + res[base + e + 256];
  red[e] = v0 + v1;
  __syncthreads();
  for (int off = 128; off > 0; off >>= 1) {
    if (e < off) red[e] += red[e + off];
    __syncthreads();
  }
  float mean = red[0] * (1.f / EMB);
  __syncthreads();
  float d0 = v0 - mean, d1 = v1 - mean;
  red[e] = d0 * d0 + d1 * d1;
  __syncthreads();
  for (int off = 128; off > 0; off >>= 1) {
    if (e < off) red[e] += red[e + off];
    __syncthreads();
  }
  float rstd = rsqrtf(red[0] * (1.f / EMB) + 1e-5f);
  float o0 = d0 * rstd * g[e] + bta[e];
  float o1 = d1 * rstd * g[e + 256] + bta[e + 256];
  if (out32) { out32[base + e] = o0; out32[base + e + 256] = o1; }
  if (outbf) { outbf[base + e] = to_bf16(o0); outbf[base + e + 256] = to_bf16(o1); }
}

// ---------------------------------------------------------------------------
// Host orchestration
// ---------------------------------------------------------------------------
extern "C" void kernel_launch(void* const* d_in, const int* in_sizes, int n_in,
                              void* d_out, int out_size, void* d_ws, size_t ws_size,
                              hipStream_t stream) {
  (void)in_sizes; (void)n_in; (void)out_size; (void)ws_size;
  const int*   x     = (const int*)d_in[0];
  const int*   mask  = (const int*)d_in[1];
  const float* Wenc  = (const float*)d_in[2];
  const float* benc  = (const float*)d_in[3];
  const float* em    = (const float*)d_in[4];
  const float* ed    = (const float*)d_in[5];
  const float* ewd   = (const float*)d_in[6];
  const float* ehr   = (const float*)d_in[7];
  const float* Wq = (const float*)d_in[8];   const float* bq = (const float*)d_in[9];
  const float* Wk = (const float*)d_in[10];  const float* bk = (const float*)d_in[11];
  const float* Wv = (const float*)d_in[12];  const float* bv = (const float*)d_in[13];
  const float* Wo = (const float*)d_in[14];  const float* bo = (const float*)d_in[15];
  const float* ln1g = (const float*)d_in[16]; const float* ln1b = (const float*)d_in[17];
  const float* ln2g = (const float*)d_in[18]; const float* ln2b = (const float*)d_in[19];
  const float* W1 = (const float*)d_in[20];  const float* b1 = (const float*)d_in[21];
  const float* W2 = (const float*)d_in[22];  const float* b2 = (const float*)d_in[23];
  float* out = (float*)d_out;

  // workspace layout (~106 MB total).
  // [64MB, 96MB) holds q/k/v/attn-out (bf16) during attention and is reused
  // as the 32MB ff1 activation afterwards (those buffers are dead by then).
  char* w = (char*)d_ws;
  const size_t MB = 1024 * 1024;
  float*  h32    = (float*) (w + 0 * MB);     // 16 MB
  __bf16* hbf    = (__bf16*)(w + 16 * MB);    //  8 MB
  float*  x1_32  = (float*) (w + 24 * MB);    // 16 MB
  __bf16* x1_bf  = (__bf16*)(w + 40 * MB);    //  8 MB
  float*  ff2_32 = (float*) (w + 48 * MB);    // 16 MB
  __bf16* qbf    = (__bf16*)(w + 64 * MB);    //  8 MB
  __bf16* kbf    = (__bf16*)(w + 72 * MB);    //  8 MB
  __bf16* vbf    = (__bf16*)(w + 80 * MB);    //  8 MB
  __bf16* atbf   = (__bf16*)(w + 88 * MB);    //  8 MB
  __bf16* ff1_bf = (__bf16*)(w + 64 * MB);    // alias of [64MB,96MB)
  __bf16* vTb    = (__bf16*)(w + 96 * MB);    //  8 MB
  __bf16* wbuf   = (__bf16*)(w + 104 * MB);   //  2 MB (transposed weight slab)

  embed_kernel<<<MTOK, 256, 0, stream>>>(x, Wenc, benc, em, ed, ewd, ehr, h32, hbf);

  dim3 gEE(EMB / 64, MTOK / 128);    // GEMM grid for Nw=512
  dim3 gEF(FFD / 64, MTOK / 128);    // GEMM grid for Nw=2048
  int tEE = (EMB * EMB + 255) / 256;
  int tEF = (EMB * FFD + 255) / 256;

  for (int i = 0; i < NLAY; ++i) {
    const float* Wq_i = Wq + (size_t)i * EMB * EMB;
    const float* Wk_i = Wk + (size_t)i * EMB * EMB;
    const float* Wv_i = Wv + (size_t)i * EMB * EMB;
    const float* Wo_i = Wo + (size_t)i * EMB * EMB;
    const float* W1_i = W1 + (size_t)i * EMB * FFD;
    const float* W2_i = W2 + (size_t)i * FFD * EMB;
    float* o_i = out + (size_t)(1 + i) * MTOK * EMB;

    // Q, K, V projections (bf16 outputs)
    transpose_bf16<<<tEE, 256, 0, stream>>>(Wq_i, wbuf, EMB, EMB);
    gemm_bf16<<<gEE, 256, 0, stream>>>(hbf, wbuf, bq + (size_t)i * EMB, nullptr, qbf, MTOK, EMB, EMB, 0);
    transpose_bf16<<<tEE, 256, 0, stream>>>(Wk_i, wbuf, EMB, EMB);
    gemm_bf16<<<gEE, 256, 0, stream>>>(hbf, wbuf, bk + (size_t)i * EMB, nullptr, kbf, MTOK, EMB, EMB, 0);
    transpose_bf16<<<tEE, 256, 0, stream>>>(Wv_i, wbuf, EMB, EMB);
    gemm_bf16<<<gEE, 256, 0, stream>>>(hbf, wbuf, bv + (size_t)i * EMB, nullptr, vbf, MTOK, EMB, EMB, 0);

    // attention
    head_transpose<<<(MTOK * EMB) / 256, 256, 0, stream>>>(vbf, vTb);
    attn_kernel<<<dim3(SEQ / 16, NHEAD, NBATCH), 32, 0, stream>>>(qbf, kbf, vTb, mask, atbf);

    // O projection straight into d_out tuple slot (fp32)
    transpose_bf16<<<tEE, 256, 0, stream>>>(Wo_i, wbuf, EMB, EMB);
    gemm_bf16<<<gEE, 256, 0, stream>>>(atbf, wbuf, bo + (size_t)i * EMB, o_i, nullptr, MTOK, EMB, EMB, 0);

    // x1 = LN(o + h)
    add_ln_kernel<<<MTOK, 256, 0, stream>>>(o_i, h32, ln1g + (size_t)i * EMB,
                                            ln1b + (size_t)i * EMB, x1_32, x1_bf);

    // FFN
    transpose_bf16<<<tEF, 256, 0, stream>>>(W1_i, wbuf, EMB, FFD);
    gemm_bf16<<<gEF, 256, 0, stream>>>(x1_bf, wbuf, b1 + (size_t)i * FFD, nullptr, ff1_bf, MTOK, EMB, FFD, 1);
    transpose_bf16<<<tEF, 256, 0, stream>>>(W2_i, wbuf, FFD, EMB);
    gemm_bf16<<<gEE, 256, 0, stream>>>(ff1_bf, wbuf, b2 + (size_t)i * EMB, ff2_32, nullptr, MTOK, FFD, EMB, 0);

    // h = LN(ff + x1); final layer writes h directly to d_out[0..M*E)
    float*  hdst  = (i == NLAY - 1) ? out : h32;
    __bf16* hbdst = (i == NLAY - 1) ? nullptr : hbf;
    add_ln_kernel<<<MTOK, 256, 0, stream>>>(ff2_32, x1_32, ln2g + (size_t)i * EMB,
                                            ln2b + (size_t)i * EMB, hdst, hbdst);
  }
}
